// TransformerBlock_Dual_72043781423218
// MI455X (gfx1250) — matrix operand order repaired and checked by reading, not compiled, not run on hardware
//
#include <hip/hip_runtime.h>
#include <math.h>

// ---------------- constants ----------------
static constexpr int  BATCH = 8;
static constexpr int  DIM   = 64;
static constexpr int  HIMG  = 128;
static constexpr int  WIMG  = 128;
static constexpr int  NPIX  = HIMG * WIMG;       // 16384
static constexpr int  PTOT  = BATCH * NPIX;      // 131072
static constexpr int  HEADS = 8;
static constexpr int  CH    = DIM / HEADS;       // 8
static constexpr int  HID   = 2 * DIM;           // 128
static constexpr int  DCONV = DIM / 4;           // 16

typedef __attribute__((ext_vector_type(16))) _Float16 v16h;
typedef __attribute__((ext_vector_type(8)))  float    v8f;

__device__ __forceinline__ float gelu_exact(float x) {
    return 0.5f * x * (1.0f + erff(x * 0.7071067811865476f));
}

// ---------------- weight f32 -> f16 ----------------
__global__ void cvt_f32_to_f16(const float* __restrict__ src,
                               _Float16* __restrict__ dst, int n) {
    int t = blockIdx.x * blockDim.x + threadIdx.x;
    if (t < n) dst[t] = (_Float16)src[t];
}

// ---------------- LN1: x[B,C,H,W] -> pixel-major f16 [P, 64] ----------------
__global__ void ln1_kernel(const float* __restrict__ x,
                           const float* __restrict__ lw,
                           const float* __restrict__ lb,
                           _Float16* __restrict__ y16) {
    int p = blockIdx.x * blockDim.x + threadIdx.x;       // < PTOT
    int b = p >> 14, n = p & (NPIX - 1);
    const float* base = x + (size_t)b * DIM * NPIX + n;
    float s = 0.f, s2 = 0.f;
    for (int c = 0; c < DIM; ++c) {
        float v = base[(size_t)c * NPIX];
        s += v; s2 += v * v;
    }
    float mu  = s * (1.0f / DIM);
    float var = s2 * (1.0f / DIM) - mu * mu;
    float rs  = rsqrtf(var + 1e-5f);
    _Float16* yp = y16 + (size_t)p * DIM;
    for (int c = 0; c < DIM; ++c) {
        float v = base[(size_t)c * NPIX];
        yp[c] = (_Float16)((v - mu) * rs * lw[c] + lb[c]);
    }
}

// ---------------- LN2: X1 CP [64,P] -> y2 CP f32 [64,P] ----------------
__global__ void ln2_kernel(const float* __restrict__ x1,
                           const float* __restrict__ lw,
                           const float* __restrict__ lb,
                           float* __restrict__ y2) {
    int p = blockIdx.x * blockDim.x + threadIdx.x;       // < PTOT
    float s = 0.f, s2 = 0.f;
    for (int c = 0; c < DIM; ++c) {
        float v = x1[(size_t)c * PTOT + p];
        s += v; s2 += v * v;
    }
    float mu  = s * (1.0f / DIM);
    float var = s2 * (1.0f / DIM) - mu * mu;
    float rs  = rsqrtf(var + 1e-5f);
    for (int c = 0; c < DIM; ++c) {
        float v = x1[(size_t)c * PTOT + p];
        y2[(size_t)c * PTOT + p] = (v - mu) * rs * lw[c] + lb[c];
    }
}

// ---------------- WMMA 1x1-conv GEMM: D[OC,P] = W[OC,IC] * Xpm[P,IC]^T ----
// MODE 0: Dst[row*P+col] = acc                         (plain CP store)
// MODE 1: Dst[row*P+col] = Aux[bchw(row,col)] + acc    (proj + x residual)
// MODE 2: Dst[bchw(row,col)] = Aux[row*P+col] + acc    (pout + X1 residual -> BCHW out)
template <int MODE>
__global__ void gemm1x1_wmma(const _Float16* __restrict__ Wh,
                             const _Float16* __restrict__ Xh,
                             float* __restrict__ Dst,
                             const float* __restrict__ Aux,
                             int OC, int IC) {
    const int lane = threadIdx.x & 31;
    const int wave = (blockIdx.x * blockDim.x + threadIdx.x) >> 5;
    const int ptiles = PTOT / 64;                        // 2048
    const int mt = wave / ptiles;
    const int pt = wave % ptiles;
    const int m0 = mt * 16;
    const int p0 = pt * 64;
    const int half = lane >> 4;
    const int l15  = lane & 15;

    v8f acc[4] = {};
    for (int k0 = 0; k0 < IC; k0 += 32) {
        // A fragment: row m0+l15, K per wave32 16-bit A layout
        v16h a;
        {
            const _Float16* ap = Wh + (size_t)(m0 + l15) * IC + k0 + half * 8;
#pragma unroll
            for (int j = 0; j < 8; ++j) a[j] = ap[j];
#pragma unroll
            for (int j = 0; j < 8; ++j) a[8 + j] = ap[16 + j];
        }
        // 4 B fragments (16 pixels each), K per wave32 16-bit layout (two 8-half runs per lane)
#pragma unroll
        for (int s = 0; s < 4; ++s) {
            int col = p0 + s * 16 + l15;
            const _Float16* bp = Xh + (size_t)col * IC + k0 + half * 8;
            v16h bf;
#pragma unroll
            for (int j = 0; j < 16; ++j) bf[j] = bp[(j & 7) + ((j >> 3) << 4)];
            acc[s] = __builtin_amdgcn_wmma_f32_16x16x32_f16(
                false, a, false, bf, (short)0, acc[s], false, false);
        }
    }
#pragma unroll
    for (int s = 0; s < 4; ++s) {
#pragma unroll
        for (int r = 0; r < 8; ++r) {
            int row = m0 + r + half * 8;
            int col = p0 + s * 16 + l15;
            size_t cp = (size_t)row * PTOT + col;
            float v = acc[s][r];
            if (MODE == 0) {
                Dst[cp] = v;
            } else {
                int b = col >> 14, n = col & (NPIX - 1);
                size_t bchw = ((size_t)b * DIM + row) * NPIX + n;
                if (MODE == 1) Dst[cp]   = Aux[bchw] + v;
                else           Dst[bchw] = Aux[cp]   + v;
            }
        }
    }
}

// ---------------- depthwise 3x3 SAME on CP planes ----------------
__global__ void dwconv3_cp(const float* __restrict__ src,
                           const float* __restrict__ wdw,
                           float* __restrict__ dst, int nch) {
    size_t t = (size_t)blockIdx.x * blockDim.x + threadIdx.x;  // < nch*PTOT
    int ch = (int)(t / PTOT);
    int p  = (int)(t - (size_t)ch * PTOT);
    int b = p >> 14, n = p & (NPIX - 1);
    int h = n >> 7, wc = n & (WIMG - 1);
    const float* sp = src + (size_t)ch * PTOT + (size_t)b * NPIX;
    const float* wp = wdw + ch * 9;
    float acc = 0.f;
#pragma unroll
    for (int ky = 0; ky < 3; ++ky) {
        int hh = h + ky - 1;
        if (hh < 0 || hh >= HIMG) continue;
#pragma unroll
        for (int kx = 0; kx < 3; ++kx) {
            int ww = wc + kx - 1;
            if (ww < 0 || ww >= WIMG) continue;
            acc += sp[hh * WIMG + ww] * wp[ky * 3 + kx];
        }
    }
    dst[t] = acc;
}

// ---------------- partial conv 3x3: 16ch -> 16ch ----------------
__global__ void pconv_kernel(const float* __restrict__ y2,
                             const float* __restrict__ pw,
                             float* __restrict__ z) {
    size_t t = (size_t)blockIdx.x * blockDim.x + threadIdx.x;  // < 16*PTOT
    int oc = (int)(t / PTOT);
    int p  = (int)(t - (size_t)oc * PTOT);
    int b = p >> 14, n = p & (NPIX - 1);
    int h = n >> 7, wc = n & (WIMG - 1);
    float acc = 0.f;
    for (int ic = 0; ic < DCONV; ++ic) {
        const float* sp = y2 + (size_t)ic * PTOT + (size_t)b * NPIX;
        const float* wp = pw + ((size_t)oc * DCONV + ic) * 9;
#pragma unroll
        for (int ky = 0; ky < 3; ++ky) {
            int hh = h + ky - 1;
            if (hh < 0 || hh >= HIMG) continue;
#pragma unroll
            for (int kx = 0; kx < 3; ++kx) {
                int ww = wc + kx - 1;
                if (ww < 0 || ww >= WIMG) continue;
                acc += sp[hh * WIMG + ww] * wp[ky * 3 + kx];
            }
        }
    }
    z[t] = acc;
}

// ---------------- pack pconv result + passthrough -> pixel-major f16 ------
__global__ void pack_z16(const float* __restrict__ z,
                         const float* __restrict__ y2,
                         _Float16* __restrict__ h16) {
    size_t t = (size_t)blockIdx.x * blockDim.x + threadIdx.x;  // < PTOT*64
    int c = (int)(t & (DIM - 1));
    size_t p = t >> 6;
    float v = (c < DCONV) ? z[(size_t)c * PTOT + p] : y2[(size_t)c * PTOT + p];
    h16[t] = (_Float16)v;
}

// ---------------- CP f32 -> pixel-major f16 ----------------
__global__ void cvt_cp_to_pm16(const float* __restrict__ src,
                               _Float16* __restrict__ dst, int C) {
    size_t t = (size_t)blockIdx.x * blockDim.x + threadIdx.x;  // < C*PTOT
    int c = (int)(t % C);
    size_t p = t / C;
    dst[t] = (_Float16)src[(size_t)c * PTOT + p];
}

// ---------------- q/k row L2 norms ----------------
__global__ void rownorm_kernel(const float* __restrict__ qkv,
                               float* __restrict__ norms) {
    __shared__ float red[256];
    int ch = blockIdx.x >> 3;            // 0..127 (q: 0..63, k: 64..127)
    int b  = blockIdx.x & 7;
    const float* base = qkv + (size_t)ch * PTOT + (size_t)b * NPIX;
    float s = 0.f;
    for (int i = threadIdx.x; i < NPIX; i += 256) { float v = base[i]; s += v * v; }
    red[threadIdx.x] = s; __syncthreads();
    for (int st = 128; st > 0; st >>= 1) {
        if ((int)threadIdx.x < st) red[threadIdx.x] += red[threadIdx.x + st];
        __syncthreads();
    }
    if (threadIdx.x == 0) norms[blockIdx.x] = fmaxf(sqrtf(red[0]), 1e-12f);
}

// ---------------- scores: one block per (b,h,c,d) ----------------
__global__ void scores_kernel(const float* __restrict__ qkv,
                              const float* __restrict__ norms,
                              float* __restrict__ S) {
    __shared__ float red[256];
    int d = blockIdx.x & 7;
    int c = (blockIdx.x >> 3) & 7;
    int h = (blockIdx.x >> 6) & 7;
    int b = blockIdx.x >> 9;
    int qch = h * CH + c;
    int kch = DIM + h * CH + d;
    const float* qp = qkv + (size_t)qch * PTOT + (size_t)b * NPIX;
    const float* kp = qkv + (size_t)kch * PTOT + (size_t)b * NPIX;
    float s = 0.f;
    for (int i = threadIdx.x; i < NPIX; i += 256) s += qp[i] * kp[i];
    red[threadIdx.x] = s; __syncthreads();
    for (int st = 128; st > 0; st >>= 1) {
        if ((int)threadIdx.x < st) red[threadIdx.x] += red[threadIdx.x + st];
        __syncthreads();
    }
    if (threadIdx.x == 0)
        S[blockIdx.x] = red[0] / (norms[qch * 8 + b] * norms[kch * 8 + b]);
}

// ---------------- softmax/relu^2 mix: one thread per (b,h,c) --------------
__global__ void attnmix_kernel(const float* __restrict__ S,
                               const float* __restrict__ temp,
                               const float* __restrict__ mix,
                               float* __restrict__ A) {
    int t = blockIdx.x * blockDim.x + threadIdx.x;   // < 512
    if (t >= BATCH * HEADS * CH) return;
    int h = (t >> 3) & 7;
    const float* sp = S + (size_t)t * CH;
    float T = temp[h];
    float m = -3.4e38f;
    for (int d = 0; d < CH; ++d) m = fmaxf(m, sp[d] * T);
    float e[CH]; float se = 0.f;
    for (int d = 0; d < CH; ++d) { e[d] = __expf(sp[d] * T - m); se += e[d]; }
    float w0 = __expf(mix[0]), w1 = __expf(mix[1]);
    float inv = 1.0f / (w0 + w1); w0 *= inv; w1 *= inv;
    for (int d = 0; d < CH; ++d) {
        float r = fmaxf(sp[d], 0.f);
        A[(size_t)t * CH + d] = w0 * e[d] / se + w1 * r * r;
    }
}

// ---------------- out = attn @ v ----------------
__global__ void attnout_kernel(const float* __restrict__ qkv,
                               const float* __restrict__ A,
                               float* __restrict__ O) {
    size_t t = (size_t)blockIdx.x * blockDim.x + threadIdx.x;  // < 64*PTOT
    int oc = (int)(t / PTOT);
    int p  = (int)(t - (size_t)oc * PTOT);
    int b = p >> 14;
    int h = oc >> 3, c = oc & 7;
    const float* ap = A + ((((size_t)b * HEADS + h) * CH) + c) * CH;
    float acc = 0.f;
#pragma unroll
    for (int d = 0; d < CH; ++d)
        acc += ap[d] * qkv[(size_t)(2 * DIM + h * CH + d) * PTOT + p];
    O[t] = acc;
}

// ---------------- gelu(a) * g, in place on a ----------------
__global__ void gelumul_kernel(float* __restrict__ a,
                               const float* __restrict__ g) {
    size_t t = (size_t)blockIdx.x * blockDim.x + threadIdx.x;  // < 128*PTOT
    a[t] = gelu_exact(a[t]) * g[t];
}

// ======================= host launcher =======================
extern "C" void kernel_launch(void* const* d_in, const int* in_sizes, int n_in,
                              void* d_out, int out_size, void* d_ws, size_t ws_size,
                              hipStream_t stream) {
    const float* x       = (const float*)d_in[0];
    const float* ln1_w   = (const float*)d_in[1];
    const float* ln1_b   = (const float*)d_in[2];
    const float* qkv_w   = (const float*)d_in[3];
    const float* qkv_dw  = (const float*)d_in[4];
    const float* temp    = (const float*)d_in[5];
    const float* amix    = (const float*)d_in[6];
    const float* proj_w  = (const float*)d_in[7];
    const float* ln2_w   = (const float*)d_in[8];
    const float* ln2_b   = (const float*)d_in[9];
    const float* pconv_w = (const float*)d_in[10];
    const float* pin_w   = (const float*)d_in[11];
    const float* dw_w    = (const float*)d_in[12];
    const float* pout_w  = (const float*)d_in[13];
    float* out = (float*)d_out;

    char* w = (char*)d_ws;
    float*    F1   = (float*)(w);                        // 256*P f32 (qkv -> O/y2 -> a|g)
    float*    F2   = (float*)(w + 134217728ull);         // 192*P f32 (qkv_dw -> z -> a_dw/u)
    float*    F3   = (float*)(w + 234881024ull);         // 64*P  f32 (X1)
    _Float16* H16  = (_Float16*)(w + 268435456ull);      // P*128 f16 (GEMM B operand)
    _Float16* Wq   = (_Float16*)(w + 301989888ull);      // 192*64
    _Float16* Wp   = (_Float16*)(w + 302014464ull);      // 64*64
    _Float16* Wpi  = (_Float16*)(w + 302022656ull);      // 256*64
    _Float16* Wpo  = (_Float16*)(w + 302055424ull);      // 64*128
    float*    norms = (float*)(w + 302071808ull);        // 1024
    float*    S     = (float*)(w + 302075904ull);        // 4096
    float*    Attn  = (float*)(w + 302092288ull);        // 4096

    const int T = 256;

    // weights -> f16
    cvt_f32_to_f16<<<(192 * 64 + T - 1) / T, T, 0, stream>>>(qkv_w, Wq, 192 * 64);
    cvt_f32_to_f16<<<(64 * 64 + T - 1) / T, T, 0, stream>>>(proj_w, Wp, 64 * 64);
    cvt_f32_to_f16<<<(256 * 64 + T - 1) / T, T, 0, stream>>>(pin_w, Wpi, 256 * 64);
    cvt_f32_to_f16<<<(64 * 128 + T - 1) / T, T, 0, stream>>>(pout_w, Wpo, 64 * 128);

    // LN1 -> pixel-major f16
    ln1_kernel<<<PTOT / T, T, 0, stream>>>(x, ln1_w, ln1_b, H16);

    // qkv 1x1 GEMM (192x64), WMMA -> F1
    gemm1x1_wmma<0><<<(192 / 16) * (PTOT / 64) / 8, T, 0, stream>>>(Wq, H16, F1, nullptr, 192, 64);

    // depthwise 3x3 on 192 planes -> F2
    dwconv3_cp<<<(size_t)192 * PTOT / T, T, 0, stream>>>(F1, qkv_dw, F2, 192);

    // attention
    rownorm_kernel<<<1024, T, 0, stream>>>(F2, norms);
    scores_kernel<<<4096, T, 0, stream>>>(F2, norms, S);
    attnmix_kernel<<<2, T, 0, stream>>>(S, temp, amix, Attn);
    attnout_kernel<<<(size_t)64 * PTOT / T, T, 0, stream>>>(F2, Attn, F1);   // O -> F1[0:64P]

    // proj GEMM (64x64) + x residual -> X1 (F3)
    cvt_cp_to_pm16<<<(size_t)64 * PTOT / T, T, 0, stream>>>(F1, H16, 64);
    gemm1x1_wmma<1><<<(64 / 16) * (PTOT / 64) / 8, T, 0, stream>>>(Wp, H16, F3, x, 64, 64);

    // FDFN
    ln2_kernel<<<PTOT / T, T, 0, stream>>>(F3, ln2_w, ln2_b, F1);             // y2 -> F1[0:64P]
    pconv_kernel<<<(size_t)16 * PTOT / T, T, 0, stream>>>(F1, pconv_w, F2);   // z  -> F2[0:16P]
    pack_z16<<<(size_t)64 * PTOT / T, T, 0, stream>>>(F2, F1, H16);
    gemm1x1_wmma<0><<<(256 / 16) * (PTOT / 64) / 8, T, 0, stream>>>(Wpi, H16, F1, nullptr, 256, 64);
    dwconv3_cp<<<(size_t)128 * PTOT / T, T, 0, stream>>>(F1, dw_w, F2, 128);  // a_dw -> F2
    gelumul_kernel<<<(size_t)128 * PTOT / T, T, 0, stream>>>(F2, F1 + (size_t)128 * PTOT);
    cvt_cp_to_pm16<<<(size_t)128 * PTOT / T, T, 0, stream>>>(F2, H16, 128);
    gemm1x1_wmma<2><<<(64 / 16) * (PTOT / 64) / 8, T, 0, stream>>>(Wpo, H16, out, F3, 64, 128);
}
